// Attention_82317343195308
// MI455X (gfx1250) — matrix operand order repaired and checked by reading, not compiled, not run on hardware
//
#include <hip/hip_runtime.h>
#include <hip/hip_bf16.h>
#include <math.h>

#define HD    512
#define SEQ   2048
#define BATCH 64
#define ROWS  64          // key rows per block
#define NTHR  128         // 4 waves
#define KPAD  520         // padded bf16 row stride (1040B): 4-bank rotation/row

typedef __bf16 bf16;
typedef __attribute__((ext_vector_type(16))) __bf16 v16bf;
typedef __attribute__((ext_vector_type(8)))  __bf16 v8bf;
typedef __attribute__((ext_vector_type(4)))  __bf16 v4bf;
typedef __attribute__((ext_vector_type(8)))  float  v8f;
typedef __attribute__((ext_vector_type(4)))  int    v4i;

#if defined(__has_builtin)
#if __has_builtin(__builtin_amdgcn_global_load_async_to_lds_b128) && \
    __has_builtin(__builtin_amdgcn_s_wait_asynccnt)
#define USE_ASYNC 1
#endif
#endif
#ifndef USE_ASYNC
#define USE_ASYNC 0
#endif

// AS-qualified element types; pointers to these are AS-qualified pointers.
typedef __attribute__((address_space(1))) v4i v4i_g;   // global
typedef __attribute__((address_space(3))) v4i v4i_l;   // LDS

__device__ __forceinline__ v16bf combine16(v8bf lo, v8bf hi) {
    v16bf r;
#pragma unroll
    for (int i = 0; i < 8; ++i) { r[i] = lo[i]; r[i + 8] = hi[i]; }
    return r;
}

// ---------------------------------------------------------------------------
// Kernel 0: Ua_w f32 -> bf16 (one-time, so GEMM staging is pure byte copies)
// ---------------------------------------------------------------------------
__global__ __launch_bounds__(256) void cvt_ua_kernel(
    const float* __restrict__ Ua_w, bf16* __restrict__ ua_bf)
{
    const int idx = blockIdx.x * 256 + threadIdx.x;   // float4 group
    float4 f = *(const float4*)(Ua_w + (size_t)idx * 4);
    v4bf o = { (bf16)f.x, (bf16)f.y, (bf16)f.z, (bf16)f.w };
    *(v4bf*)(ua_bf + (size_t)idx * 4) = o;
}

// ---------------------------------------------------------------------------
// Kernel 1: qp[b][d] = sum_h query[b][h]*Wa_w[d][h] + Wa_b[d] + Ua_b[d]
// ---------------------------------------------------------------------------
__global__ __launch_bounds__(256) void qproj_kernel(
    const float* __restrict__ query, const float* __restrict__ Wa_w,
    const float* __restrict__ Wa_b,  const float* __restrict__ Ua_b,
    float* __restrict__ qp)
{
    const int b = blockIdx.x;
    const int tid = threadIdx.x;
    __shared__ float qS[HD];
    for (int h = tid; h < HD; h += 256) qS[h] = query[b * HD + h];
    __syncthreads();
    for (int d = tid; d < HD; d += 256) {
        const float* w = Wa_w + (size_t)d * HD;
        float acc = 0.f;
#pragma unroll 4
        for (int h = 0; h < HD; h += 4) {
            acc += qS[h]     * w[h];
            acc += qS[h + 1] * w[h + 1];
            acc += qS[h + 2] * w[h + 2];
            acc += qS[h + 3] * w[h + 3];
        }
        qp[b * HD + d] = acc + Wa_b[d] + Ua_b[d];
    }
}

// ---------------------------------------------------------------------------
// Kernel 2 (WMMA): raw scores. Block = (batch, 64 key rows), 4 waves.
// ---------------------------------------------------------------------------
#if USE_ASYNC
__device__ __forceinline__ void stage_ua_async(
    const bf16* __restrict__ ua_bf, bf16 (*uaT)[KPAD], int dBase, int tid)
{
    // 16 x 512 bf16 = 16KB, 128 lanes * 8 issues * 16B each
#pragma unroll
    for (int i = 0; i < 8; ++i) {
        const int seg = tid + (i << 7);       // 0..1023
        const int dl  = seg >> 6;             // 0..15
        const int hs  = (seg & 63) << 3;      // bf16 index, 16B granules
        v4i_g* gsrc = (v4i_g*)(uintptr_t)
                      (ua_bf + (size_t)(dBase + dl) * HD + hs);
        // low 32 bits of a flat LDS address == wave-relative LDS offset
        v4i_l* ldst = (v4i_l*)(unsigned)(uintptr_t)(&uaT[dl][hs]);
        __builtin_amdgcn_global_load_async_to_lds_b128(gsrc, ldst, 0, 0);
    }
}
#endif

__global__ __launch_bounds__(NTHR) void scores_kernel(
    const float* __restrict__ keys, const float* __restrict__ qp,
    const bf16* __restrict__ ua_bf, const float* __restrict__ Va_w,
    const float* __restrict__ Va_b, float* __restrict__ scoresRaw)
{
    const int b     = blockIdx.y;
    const int sBase = blockIdx.x * ROWS;
    const int tid   = threadIdx.x;        // 0..127
    const int lane  = tid & 31;
    const int wave  = tid >> 5;           // 0..3
    const int half  = lane >> 4;
    const int col   = lane & 15;
    const int row   = (wave << 4) + col;  // A row within 64-row tile

    __shared__ bf16 keysT[ROWS][KPAD];    // 65 KB
    __shared__ bf16 uaT[2][16][KPAD];     // 32.5 KB (double buffer)

    // ---- stage keys tile f32 -> bf16 (float4 loads, packed bf16 stores) ----
    const float* kbase = keys + ((size_t)b * SEQ + sBase) * HD;
    for (int idx = tid; idx < (ROWS * HD) / 4; idx += NTHR) {
        const int r = idx >> 7;               // HD/4 = 128 groups per row
        const int g = (idx & 127) << 2;
        float4 f = *(const float4*)(kbase + (size_t)r * HD + g);
        v4bf o = { (bf16)f.x, (bf16)f.y, (bf16)f.z, (bf16)f.w };
        *(v4bf*)&keysT[r][g] = o;
    }

    float srow[8];
#pragma unroll
    for (int i = 0; i < 8; ++i) srow[i] = 0.f;

    const int aOff = half ? 8 : 0;
    const int bOff = half ? 16 : 0;

#if USE_ASYNC
    stage_ua_async(ua_bf, uaT[0], 0, tid);
    __builtin_amdgcn_s_wait_asynccnt(0);
    __syncthreads();                      // keysT + uaT[0] ready
#endif

    for (int dc = 0; dc < HD / 16; ++dc) {
        const int dBase = dc << 4;
#if USE_ASYNC
        const int cur = dc & 1;
        if (dc + 1 < HD / 16)
            stage_ua_async(ua_bf, uaT[cur ^ 1], dBase + 16, tid);
        const bf16 (*ua)[KPAD] = uaT[cur];
#else
        __syncthreads();                  // prior-iter reads done
        for (int i = 0; i < 8; ++i) {     // manual 16B copies
            const int seg = tid + (i << 7);
            const int dl  = seg >> 6;
            const int hs  = (seg & 63) << 3;
            *(v8bf*)&uaT[0][dl][hs] =
                *(const v8bf*)(ua_bf + (size_t)(dBase + dl) * HD + hs);
        }
        __syncthreads();
        const bf16 (*ua)[KPAD] = uaT[0];
#endif

        v8f c0 = {}, c1 = {};             // two independent WMMA chains
#pragma unroll
        for (int k0 = 0; k0 < HD; k0 += 64) {
            {
                v8bf a_lo = *(const v8bf*)&keysT[row][k0 + aOff];
                v8bf a_hi = *(const v8bf*)&keysT[row][k0 + 16 + aOff];
                v8bf b_lo = *(const v8bf*)&ua[col][k0 + aOff];
                v8bf b_hi = *(const v8bf*)&ua[col][k0 + 16 + aOff];
                c0 = __builtin_amdgcn_wmma_f32_16x16x32_bf16(
                        false, combine16(a_lo, a_hi), false,
                        combine16(b_lo, b_hi), (short)0, c0, false, false);
            }
            {
                const int k1 = k0 + 32;
                v8bf a_lo = *(const v8bf*)&keysT[row][k1 + aOff];
                v8bf a_hi = *(const v8bf*)&keysT[row][k1 + 16 + aOff];
                v8bf b_lo = *(const v8bf*)&ua[col][k1 + aOff];
                v8bf b_hi = *(const v8bf*)&ua[col][k1 + 16 + aOff];
                c1 = __builtin_amdgcn_wmma_f32_16x16x32_bf16(
                        false, combine16(a_lo, a_hi), false,
                        combine16(b_lo, b_hi), (short)0, c1, false, false);
            }
        }

        const float qv = qp[b * HD + dBase + col];
        const float vv = Va_w[dBase + col];
#pragma unroll
        for (int i = 0; i < 8; ++i)
            srow[i] += tanhf(c0[i] + c1[i] + qv) * vv;

#if USE_ASYNC
        __builtin_amdgcn_s_wait_asynccnt(0);
        __syncthreads();                  // next buffer filled, all reads done
#endif
    }

    // reduce the 16 d-columns (lanes 0-15 / 16-31 independently)
#pragma unroll
    for (int i = 0; i < 8; ++i) {
        float v = srow[i];
        v += __shfl_xor(v, 8, 16);
        v += __shfl_xor(v, 4, 16);
        v += __shfl_xor(v, 2, 16);
        v += __shfl_xor(v, 1, 16);
        srow[i] = v;
    }
    if ((lane & 15) == 0) {
        const int mBase = (wave << 4) + (half << 3);
        const float vb = Va_b[0];
#pragma unroll
        for (int i = 0; i < 8; ++i)
            scoresRaw[(size_t)b * SEQ + sBase + mBase + i] = srow[i] + vb;
    }
}

// ---------------------------------------------------------------------------
// Kernel 3: softmax over S per batch
// ---------------------------------------------------------------------------
__global__ __launch_bounds__(256) void softmax_kernel(
    const float* __restrict__ scoresRaw, float* __restrict__ attn)
{
    const int b = blockIdx.x;
    const int tid = threadIdx.x;
    __shared__ float red[256];

    float m = -3.402823466e+38f;
    for (int s = tid; s < SEQ; s += 256)
        m = fmaxf(m, scoresRaw[(size_t)b * SEQ + s]);
    red[tid] = m; __syncthreads();
    for (int off = 128; off > 0; off >>= 1) {
        if (tid < off) red[tid] = fmaxf(red[tid], red[tid + off]);
        __syncthreads();
    }
    const float mx = red[0];
    __syncthreads();

    float sum = 0.f;
    for (int s = tid; s < SEQ; s += 256) {
        const float e = __expf(scoresRaw[(size_t)b * SEQ + s] - mx);
        attn[(size_t)b * SEQ + s] = e;
        sum += e;
    }
    red[tid] = sum; __syncthreads();
    for (int off = 128; off > 0; off >>= 1) {
        if (tid < off) red[tid] += red[tid + off];
        __syncthreads();
    }
    const float inv = 1.0f / red[0];
    __syncthreads();

    for (int s = tid; s < SEQ; s += 256)
        attn[(size_t)b * SEQ + s] *= inv;
}

// ---------------------------------------------------------------------------
// Kernel 4: context[b][h] = sum_s attn[b][s] * keys[b][s][h]
// ---------------------------------------------------------------------------
__global__ __launch_bounds__(256) void context_kernel(
    const float* __restrict__ attn, const float* __restrict__ keys,
    float* __restrict__ ctx)
{
    const int b = blockIdx.x;
    const int tid = threadIdx.x;
    __shared__ float aS[SEQ];
    for (int s = tid; s < SEQ; s += 256)
        aS[s] = attn[(size_t)b * SEQ + s];
    __syncthreads();

    const float* kb = keys + (size_t)b * SEQ * HD;
    float acc0 = 0.f, acc1 = 0.f;
#pragma unroll 4
    for (int s = 0; s < SEQ; ++s) {
        const float a = aS[s];
        acc0 += a * kb[(size_t)s * HD + tid];
        acc1 += a * kb[(size_t)s * HD + tid + 256];
    }
    ctx[(size_t)b * HD + tid]       = acc0;
    ctx[(size_t)b * HD + tid + 256] = acc1;
}

// ---------------------------------------------------------------------------
extern "C" void kernel_launch(void* const* d_in, const int* in_sizes, int n_in,
                              void* d_out, int out_size, void* d_ws, size_t ws_size,
                              hipStream_t stream)
{
    const float* query = (const float*)d_in[0];
    const float* keys  = (const float*)d_in[1];
    const float* Wa_w  = (const float*)d_in[2];
    const float* Wa_b  = (const float*)d_in[3];
    const float* Ua_w  = (const float*)d_in[4];
    const float* Ua_b  = (const float*)d_in[5];
    const float* Va_w  = (const float*)d_in[6];
    const float* Va_b  = (const float*)d_in[7];

    float* out  = (float*)d_out;
    float* ctx  = out;                        // [64, 512]
    float* attn = out + BATCH * HD;           // [64, 2048]

    float* qp        = (float*)d_ws;          // 32768 f32
    float* scoresRaw = qp + BATCH * HD;       // 131072 f32
    bf16*  ua_bf     = (bf16*)(scoresRaw + BATCH * SEQ); // 262144 bf16

    cvt_ua_kernel<<<(HD * HD) / (256 * 4), 256, 0, stream>>>(Ua_w, ua_bf);
    qproj_kernel<<<BATCH, 256, 0, stream>>>(query, Wa_w, Wa_b, Ua_b, qp);

    dim3 grid(SEQ / ROWS, BATCH);
    scores_kernel<<<grid, NTHR, 0, stream>>>(keys, qp, ua_bf, Va_w, Va_b, scoresRaw);

    softmax_kernel<<<BATCH, 256, 0, stream>>>(scoresRaw, attn);
    context_kernel<<<BATCH, 256, 0, stream>>>(attn, keys, ctx);
}
